// wavenet_dann_38946763440816
// MI455X (gfx1250) — matrix operand order repaired and checked by reading, not compiled, not run on hardware
//
#include <hip/hip_runtime.h>
#include <hip/hip_bf16.h>
#include <math.h>

// ---------------------------------------------------------------------------
// WaveNet forward, MI455X (gfx1250, wave32, WMMA bf16->f32).
// - All GEMMs -> v_wmma_f32_16x16x32_bf16 (fp32 accumulate).
// - Causal receptive-field trimming: only columns reaching the last-24 head.
// - Conv activation tiles staged to LDS via async-to-LDS (ASYNCcnt) when the
//   toolchain exposes the builtin; plain LDS staging otherwise.
// ---------------------------------------------------------------------------

typedef __bf16 bf16_t;
typedef __bf16 v16bf __attribute__((ext_vector_type(16)));
typedef float  v8f   __attribute__((ext_vector_type(8)));
typedef int    v4i   __attribute__((ext_vector_type(4)));

#define T_LEN   1024
#define NB      64
#define DHID    256
#define NLAY    8
#define TAIL    24
#define SKIP_T0 992   // first t held in the 32-column skip buffer
#define LPAD    8     // LDS row padding (elements) -> conflict-free b128 access

#define AS1 __attribute__((address_space(1)))
#define AS3 __attribute__((address_space(3)))

#if defined(__gfx1250__) && __has_builtin(__builtin_amdgcn_global_load_async_to_lds_b128)
#define HAVE_ASYNC_LDS 1
#else
#define HAVE_ASYNC_LDS 0
#endif

union BF16Frag  { v16bf v; uint4 q[2]; };
union BF16Pack8 { uint4 q; bf16_t e[8]; };

__device__ __forceinline__ v16bf ld_frag(const bf16_t* p0, const bf16_t* p1) {
  BF16Frag f;
  f.q[0] = *(const uint4*)p0;   // 8 bf16 = 16B
  f.q[1] = *(const uint4*)p1;   // 8 bf16 = 16B
  return f.v;
}
__device__ __forceinline__ v8f vzero8() {
  v8f z;
#pragma unroll
  for (int i = 0; i < 8; ++i) z[i] = 0.0f;
  return z;
}
__device__ __forceinline__ v8f wmma_bf16(v16bf a, v16bf b, v8f c) {
  // D = A(16x32) * B(32x16) + C, f32 accumulate
  return __builtin_amdgcn_wmma_f32_16x16x32_bf16(false, a, false, b,
                                                 (short)0, c, false, false);
}

// ---------------------------------------------------------------------------
// Weight prep: fp32 -> bf16.  conv re-laid-out [layer][k][o][c] (c contiguous)
// so each lane's A-fragment is two contiguous 16B loads.
// ---------------------------------------------------------------------------
__global__ void prep_conv_bf16(const float* __restrict__ w, bf16_t* __restrict__ out, int n) {
  int idx = blockIdx.x * 256 + threadIdx.x;
  if (idx >= n) return;
  int c = idx & 255;
  int o = (idx >> 8) & 511;
  int r = idx >> 17;          // r = i*3 + k
  int k = r % 3;
  int i = r / 3;
  out[idx] = (bf16_t)w[(((size_t)(i * 512 + o) * 256 + c) * 3) + k];
}

__global__ void prep_skip_bf16(const float* __restrict__ w, bf16_t* __restrict__ out, int n) {
  int idx = blockIdx.x * 256 + threadIdx.x;
  if (idx >= n) return;
  out[idx] = (bf16_t)w[idx];  // already [layer][o][c]
}

// ---------------------------------------------------------------------------
// Up-projection: h0[b,t,c] = concat(Xlag, Xcov) @ up_w + up_b, t >= tstart.
// ---------------------------------------------------------------------------
__global__ void up_kernel(const float* __restrict__ Xcov, const float* __restrict__ Xlag,
                          const float* __restrict__ upw,  const float* __restrict__ upb,
                          float* __restrict__ h32, bf16_t* __restrict__ hb, int tstart) {
  int t = tstart + blockIdx.x;
  int b = blockIdx.y;
  int c = threadIdx.x;
  __shared__ float x[16];
  if (threadIdx.x < 8)
    x[threadIdx.x] = Xlag[((size_t)t * NB + b) * 8 + threadIdx.x];
  else if (threadIdx.x < 16)
    x[threadIdx.x] = Xcov[((size_t)t * NB + b) * 8 + (threadIdx.x - 8)];
  __syncthreads();
  float acc = upb[c];
#pragma unroll
  for (int d = 0; d < 16; ++d) acc += x[d] * upw[d * DHID + c];
  size_t o = ((size_t)b * T_LEN + t) * DHID + c;
  h32[o] = acc;
  hb[o]  = (bf16_t)acc;
}

// ---------------------------------------------------------------------------
// Fused layer: stage 3 activation tap-tiles to LDS (async) -> dilated-conv
// WMMA -> gate -> skip-GEMM WMMA (B from LDS) -> residual h (+ tail skip).
// Block = 512 threads = 16 waves; one 16-column tile of one batch element.
// ---------------------------------------------------------------------------
__global__ __launch_bounds__(512) void layer_kernel(
    const bf16_t* __restrict__ hbIn, const float* __restrict__ h32In,
    bf16_t* __restrict__ hbOut, float* __restrict__ h32Out,
    float* __restrict__ skip32,
    const bf16_t* __restrict__ Wc, const bf16_t* __restrict__ Ws,
    const float* __restrict__ convb, const float* __restrict__ skipb,
    int layer, int dil, int tileBase) {

  const int b    = blockIdx.y;
  const int t0   = (tileBase + blockIdx.x) * 16;
  const int lane = threadIdx.x & 31;
  const int wv   = threadIdx.x >> 5;   // wave id 0..15
  const int half = lane >> 4;          // 0 | 1
  const int nn   = lane & 15;          // column / row-in-tile index
  const int t    = t0 + nn;            // this lane's output timestep

  __shared__ __align__(16) bf16_t ldsB[3][16][DHID + LPAD]; // 3 tap tiles
  __shared__ __align__(16) bf16_t ldsG[16][DHID + LPAD];    // gated tile

  // ---- stage the 3 causal tap tiles (16 t x 256 c bf16 each) into LDS ----
  {
    const int row = threadIdx.x >> 5;   // 0..15 (t within tile)
    const int ch  = threadIdx.x & 31;   // 0..31 (16B chunk within row)
#pragma unroll
    for (int j = 0; j < 3; ++j) {
      int tb = t0 + row - (2 - j) * dil;   // >= 448 by trimming
      if (tb < 0) tb = 0;                  // safety only, unreachable
      const bf16_t* gp = hbIn + ((size_t)b * T_LEN + tb) * DHID + ch * 8;
      bf16_t* lp = &ldsB[j][row][ch * 8];
#if HAVE_ASYNC_LDS
      __builtin_amdgcn_global_load_async_to_lds_b128(
          (AS1 v4i*)gp, (AS3 v4i*)lp, 0, 0);
#else
      *(uint4*)lp = *(const uint4*)gp;
#endif
    }
#if HAVE_ASYNC_LDS
#if __has_builtin(__builtin_amdgcn_s_wait_asynccnt)
    __builtin_amdgcn_s_wait_asynccnt(0);
#else
    asm volatile("s_wait_asynccnt 0" ::: "memory");
#endif
#endif
  }
  __syncthreads();

  // ---- f/g = bias + sum_k  W_k(512x256) @ h(:, t-(2-k)*dil) ----
  v8f facc = vzero8();
  v8f gacc = vzero8();
  const bf16_t* Wl = Wc + (size_t)layer * 3 * 512 * DHID;
#pragma unroll
  for (int k = 0; k < 3; ++k) {
    const bf16_t* Wk = Wl + (size_t)k * 512 * DHID;
    const bf16_t* aF = Wk + (size_t)(16 * wv + nn) * DHID + 8 * half;
    const bf16_t* aG = Wk + (size_t)(16 * wv + 256 + nn) * DHID + 8 * half;
#pragma unroll
    for (int ks = 0; ks < 8; ++ks) {                  // K = 256 in steps of 32
      const int c0 = ks * 32;
      const bf16_t* gp = &ldsB[k][nn][c0 + 8 * half];
      BF16Frag bfr;
      bfr.q[0] = *(const uint4*)gp;        // ds_load_b128, conflict-free
      bfr.q[1] = *(const uint4*)(gp + 16);
      v16bf af = ld_frag(aF + c0, aF + c0 + 16);
      v16bf ag = ld_frag(aG + c0, aG + c0 + 16);
      facc = wmma_bf16(af, bfr.v, facc);
      gacc = wmma_bf16(ag, bfr.v, gacc);
    }
  }

  const int cb = 16 * wv + 8 * half;   // channel base of this lane's 8 rows
  {
    BF16Pack8 p;
#pragma unroll
    for (int r = 0; r < 8; ++r) {
      float f  = facc[r] + convb[layer * 512 + cb + r];
      float g  = gacc[r] + convb[layer * 512 + 256 + cb + r];
      float gv = tanhf(f) * (1.0f / (1.0f + expf(-g)));
      p.e[r] = (bf16_t)gv;
    }
    *(uint4*)&ldsG[nn][cb] = p.q;      // conflict-free b128 store
  }
  __syncthreads();

  // ---- hs = skip_w(512x256) @ gated ; bottom half only for tail tiles ----
  const bool doSkip = (t0 >= SKIP_T0);               // uniform per block
  v8f hacc = vzero8();
  v8f sacc = vzero8();
  const bf16_t* Wsl = Ws + (size_t)layer * 512 * DHID;
  const bf16_t* aH  = Wsl + (size_t)(16 * wv + nn) * DHID + 8 * half;
  const bf16_t* aS  = Wsl + (size_t)(16 * wv + 256 + nn) * DHID + 8 * half;
  if (doSkip) {
#pragma unroll
    for (int ks = 0; ks < 8; ++ks) {
      const int c0 = ks * 32;
      const bf16_t* gp = &ldsG[nn][c0 + 8 * half];
      BF16Frag bfr;
      bfr.q[0] = *(const uint4*)gp;
      bfr.q[1] = *(const uint4*)(gp + 16);
      hacc = wmma_bf16(ld_frag(aH + c0, aH + c0 + 16), bfr.v, hacc);
      sacc = wmma_bf16(ld_frag(aS + c0, aS + c0 + 16), bfr.v, sacc);
    }
  } else {
#pragma unroll
    for (int ks = 0; ks < 8; ++ks) {
      const int c0 = ks * 32;
      const bf16_t* gp = &ldsG[nn][c0 + 8 * half];
      BF16Frag bfr;
      bfr.q[0] = *(const uint4*)gp;
      bfr.q[1] = *(const uint4*)(gp + 16);
      hacc = wmma_bf16(ld_frag(aH + c0, aH + c0 + 16), bfr.v, hacc);
    }
  }

  // ---- epilogue: residual h (fp32 master + bf16 copy), tail skip accum ----
  const size_t base = ((size_t)b * T_LEN + t) * DHID + cb;
  float4 h0a = *(const float4*)(h32In + base);
  float4 h0b = *(const float4*)(h32In + base + 4);
  float hv[8] = {h0a.x, h0a.y, h0a.z, h0a.w, h0b.x, h0b.y, h0b.z, h0b.w};
  BF16Pack8 pb;
#pragma unroll
  for (int r = 0; r < 8; ++r) {
    hv[r] += hacc[r] + skipb[layer * 512 + cb + r];
    pb.e[r] = (bf16_t)hv[r];
  }
  *(float4*)(h32Out + base)     = make_float4(hv[0], hv[1], hv[2], hv[3]);
  *(float4*)(h32Out + base + 4) = make_float4(hv[4], hv[5], hv[6], hv[7]);
  *(uint4*)(hbOut + base)       = pb.q;

  if (doSkip) {
    const size_t sb = ((size_t)b * 32 + (t - SKIP_T0)) * DHID + cb;
#pragma unroll
    for (int r = 0; r < 8; ++r) {
      float sv = sacc[r] + skipb[layer * 512 + 256 + cb + r];
      if (layer == 0) skip32[sb + r] = sv;      // initializes (skip starts at 0)
      else            skip32[sb + r] += sv;
    }
  }
}

// ---------------------------------------------------------------------------
// Head: ff = relu(skip @ fc_w + fc_b); loc/scale/NLL on last 24 timesteps.
// ---------------------------------------------------------------------------
__global__ void head_fc_kernel(const float* __restrict__ skip32,
                               const float* __restrict__ fcw, const float* __restrict__ fcb,
                               const float* __restrict__ locw, const float* __restrict__ locb,
                               const float* __restrict__ sclw, const float* __restrict__ sclb,
                               const float* __restrict__ y,
                               float* __restrict__ out, float* __restrict__ logp) {
  const int rr = blockIdx.x;            // rr = tt*64 + b  (matches (24,B,1) flat)
  const int tt = rr >> 6;
  const int b  = rr & 63;
  const int t  = (T_LEN - TAIL) + tt;   // 1000..1023
  const int o  = threadIdx.x;
  __shared__ float srow[DHID];
  __shared__ float r1[DHID];
  __shared__ float r2[DHID];
  srow[o] = skip32[((size_t)b * 32 + (t - SKIP_T0)) * DHID + o];
  __syncthreads();
  float acc = fcb[o];
#pragma unroll 4
  for (int c = 0; c < DHID; ++c) acc += srow[c] * fcw[c * DHID + o];
  float ff = fmaxf(acc, 0.0f);
  r1[o] = ff * locw[o];
  r2[o] = ff * sclw[o];
  __syncthreads();
  for (int s = 128; s > 0; s >>= 1) {
    if (o < s) { r1[o] += r1[o + s]; r2[o] += r2[o + s]; }
    __syncthreads();
  }
  if (o == 0) {
    float loc   = r1[0] + locb[0];
    float sr    = r2[0] + sclb[0];
    float scale = (sr > 20.0f ? sr : log1pf(expf(sr))) + 1e-6f;
    float yt    = y[(size_t)t * NB + b];
    float z     = (yt - loc) / scale;
    float lp    = -0.5f * z * z - logf(scale) - 0.91893853320467274f; // 0.5*ln(2pi)
    out[2 + rr]        = loc;
    out[2 + 1536 + rr] = scale;
    logp[rr]           = lp;
  }
}

__global__ void head_reduce_kernel(const float* __restrict__ logp, float* __restrict__ out) {
  __shared__ float r[256];
  float s = 0.0f;
  for (int i = threadIdx.x; i < TAIL * NB; i += 256) s += logp[i];
  r[threadIdx.x] = s;
  __syncthreads();
  for (int st = 128; st > 0; st >>= 1) {
    if (threadIdx.x < st) r[threadIdx.x] += r[threadIdx.x + st];
    __syncthreads();
  }
  if (threadIdx.x == 0) {
    float loss = -(r[0] / (float)(TAIL * NB));
    out[0] = loss;   // total_loss
    out[1] = loss;   // loss_y
  }
}

// ---------------------------------------------------------------------------
extern "C" void kernel_launch(void* const* d_in, const int* in_sizes, int n_in,
                              void* d_out, int out_size, void* d_ws, size_t ws_size,
                              hipStream_t stream) {
  const float* Xcov  = (const float*)d_in[1];
  const float* Xlag  = (const float*)d_in[2];
  const float* y     = (const float*)d_in[3];
  const float* upw   = (const float*)d_in[5];
  const float* upb   = (const float*)d_in[6];
  const float* convw = (const float*)d_in[7];
  const float* convb = (const float*)d_in[8];
  const float* skw   = (const float*)d_in[9];
  const float* skb   = (const float*)d_in[10];
  const float* fcw   = (const float*)d_in[11];
  const float* fcb   = (const float*)d_in[12];
  const float* locw  = (const float*)d_in[13];
  const float* locb  = (const float*)d_in[14];
  const float* sclw  = (const float*)d_in[15];
  const float* sclb  = (const float*)d_in[16];

  // workspace layout
  char* ws = (char*)d_ws;
  const size_t H32 = (size_t)NB * T_LEN * DHID * sizeof(float);   // 64 MiB
  const size_t HBF = (size_t)NB * T_LEN * DHID * sizeof(bf16_t);  // 32 MiB
  const size_t SK  = (size_t)NB * 32 * DHID * sizeof(float);      //  2 MiB
  const size_t WC  = (size_t)NLAY * 3 * 512 * DHID * sizeof(bf16_t);
  const size_t WSZ = (size_t)NLAY * 512 * DHID * sizeof(bf16_t);
  float*  h32A   = (float*)(ws);
  float*  h32B   = (float*)(ws + H32);
  bf16_t* hbA    = (bf16_t*)(ws + 2 * H32);
  bf16_t* hbB    = (bf16_t*)(ws + 2 * H32 + HBF);
  float*  skip32 = (float*)(ws + 2 * H32 + 2 * HBF);
  bf16_t* WcB    = (bf16_t*)(ws + 2 * H32 + 2 * HBF + SK);
  bf16_t* WsB    = (bf16_t*)(ws + 2 * H32 + 2 * HBF + SK + WC);
  float*  logp   = (float*)(ws + 2 * H32 + 2 * HBF + SK + WC + WSZ);

  // weight prep (runs every call; cheap, keeps launch deterministic/stateless)
  const int nconv = NLAY * 3 * 512 * DHID;
  const int nskip = NLAY * 512 * DHID;
  prep_conv_bf16<<<(nconv + 255) / 256, 256, 0, stream>>>(convw, WcB, nconv);
  prep_skip_bf16<<<(nskip + 255) / 256, 256, 0, stream>>>(skw, WsB, nskip);

  // Receptive-field trimmed tile starts (t0/16) for h^(i), i=0..8:
  // only columns that can reach the last-24 head are ever produced.
  const int Pt[9] = {28, 29, 30, 31, 32, 34, 38, 46, 62};

  const int tstart = Pt[0] * 16;  // 448
  up_kernel<<<dim3(T_LEN - tstart, NB), DHID, 0, stream>>>(Xcov, Xlag, upw, upb,
                                                           h32A, hbA, tstart);

  for (int i = 0; i < NLAY; ++i) {
    const bf16_t* hin   = (i & 1) ? hbB : hbA;
    const float*  h32in = (i & 1) ? h32B : h32A;
    bf16_t*       hout  = (i & 1) ? hbA : hbB;
    float*        h32o  = (i & 1) ? h32A : h32B;
    const int tiles = 64 - Pt[i + 1];
    layer_kernel<<<dim3(tiles, NB), 512, 0, stream>>>(
        hin, h32in, hout, h32o, skip32, WcB, WsB, convb, skb, i, 1 << i, Pt[i + 1]);
  }

  head_fc_kernel<<<TAIL * NB, DHID, 0, stream>>>(skip32, fcw, fcb, locw, locb,
                                                 sclw, sclb, y, (float*)d_out, logp);
  head_reduce_kernel<<<1, 256, 0, stream>>>(logp, (float*)d_out);

  (void)in_sizes; (void)n_in; (void)out_size; (void)ws_size;
}
